// ChainOfThoughtProcessor_48911087567715
// MI455X (gfx1250) — compile-verified, matrix operand order checked by reading, not run on hardware
//
#include <hip/hip_runtime.h>

#define Hdim 512
#define HD 64
#define SEGL 16
#define CORE 4096                 // core tokens per batch
#define MTOK (16 * 4096)          // 65536 core tokens total
#define MTOT (16 * 4097)          // 65552 tokens incl. trailing token per batch

typedef __attribute__((ext_vector_type(16))) __bf16        bf16x16;
typedef __attribute__((ext_vector_type(8)))  float         f32x8;
typedef __attribute__((ext_vector_type(4)))  unsigned int  u32x4;
typedef __attribute__((ext_vector_type(4)))  float         f32x4;

union B16Frag { bf16x16 v; u32x4 q[2]; };

__device__ __forceinline__ bf16x16 ldfrag(const void* p0, const void* p1) {
  B16Frag f;
  f.q[0] = *reinterpret_cast<const u32x4*>(p0);
  f.q[1] = *reinterpret_cast<const u32x4*>(p1);
  return f.v;
}
__device__ __forceinline__ bf16x16 zfrag() {
  B16Frag f; f.q[0] = (u32x4)0u; f.q[1] = (u32x4)0u; return f.v;
}
__device__ __forceinline__ f32x8 wmma_bf16(bf16x16 a, bf16x16 b, f32x8 c) {
  return __builtin_amdgcn_wmma_f32_16x16x32_bf16(false, a, false, b, (short)0, c, false, false);
}

// Async global->LDS copy, 16B per lane (512B per wave-op). LDS destination is
// the low 32 bits of the generic pointer (LDS aperture lives in the high half;
// addr[31:0] is the in-LDS byte offset). Tracked by ASYNCcnt.
__device__ __forceinline__ void async_copy_b128(void* lds_ptr, const void* gptr) {
  unsigned lo = (unsigned)(uintptr_t)lds_ptr;
  unsigned long long ga = (unsigned long long)(uintptr_t)gptr;
  asm volatile("global_load_async_to_lds_b128 %0, %1, off"
               :: "v"(lo), "v"(ga) : "memory");
}
__device__ __forceinline__ void wait_async_le3() {
  asm volatile("s_wait_asynccnt 0x3" ::: "memory");
}
__device__ __forceinline__ void wait_async_0() {
  asm volatile("s_wait_asynccnt 0x0" ::: "memory");
}

// ---------------------------------------------------------------------------
// Weight transpose + fp32->bf16:  Wt[n][k] = W[k][n]
// ---------------------------------------------------------------------------
__global__ void k_transpose_w(const float* __restrict__ W, __bf16* __restrict__ Wt,
                              int K, int N) {
  int idx = blockIdx.x * 256 + threadIdx.x;
  if (idx < K * N) {
    int k = idx / N, n = idx % N;
    Wt[(size_t)n * K + k] = (__bf16)W[idx];
  }
}

// ---------------------------------------------------------------------------
// Core hidden -> bf16, dropping the trailing token of each batch (row remap).
// ---------------------------------------------------------------------------
__global__ void k_convert_hidden(const float* __restrict__ hidden, __bf16* __restrict__ X) {
  size_t e = ((size_t)blockIdx.x * 256 + threadIdx.x) * 4;
  if (e >= (size_t)MTOK * Hdim) return;
  size_t row = e >> 9;
  size_t col = e & 511;
  size_t hrow = row + (row >> 12);                // b*4097 + i
  const f32x4 v = *reinterpret_cast<const f32x4*>(hidden + hrow * Hdim + col);
  __bf16* o = X + row * Hdim + col;
  o[0] = (__bf16)v[0]; o[1] = (__bf16)v[1]; o[2] = (__bf16)v[2]; o[3] = (__bf16)v[3];
}

// ---------------------------------------------------------------------------
// Marker token QKV (2 rows shared by all segments):  m{q,k,v}[t][n]
// ---------------------------------------------------------------------------
__global__ void k_marker_qkv(const float* __restrict__ s0, const float* __restrict__ s1,
                             const float* __restrict__ Wq, const float* __restrict__ bq,
                             const float* __restrict__ Wk, const float* __restrict__ bk,
                             const float* __restrict__ Wv, const float* __restrict__ bv,
                             __bf16* __restrict__ mq, __bf16* __restrict__ mk,
                             __bf16* __restrict__ mv) {
  int idx = blockIdx.x * 256 + threadIdx.x;              // 3*2*512 = 3072
  if (idx >= 3072) return;
  int which = idx / 1024;
  int t = (idx >> 9) & 1;
  int n = idx & 511;
  const float* W = (which == 0) ? Wq : (which == 1) ? Wk : Wv;
  const float* b = (which == 0) ? bq : (which == 1) ? bk : bv;
  const float* x = t ? s1 : s0;
  float acc = b[n];
  for (int k = 0; k < Hdim; ++k) acc += x[k] * W[(size_t)k * Hdim + n];
  __bf16* o = (which == 0) ? mq : (which == 1) ? mk : mv;
  o[t * Hdim + n] = (__bf16)acc;
}

// ---------------------------------------------------------------------------
// GEMM, bf16 A, double-buffered async global->LDS staging.
// Block tile 128x64, 8 waves, each 32x32 (2x2 WMMA frags), K-step 32.
// mode 0: fp32 store; 1: GELU->bf16; 2: +res->fp32; 3: plain bf16 store.
// ---------------------------------------------------------------------------
__global__ __launch_bounds__(256) void k_gemm_bf16(
    const __bf16* __restrict__ A, int M, int Kd,
    const __bf16* __restrict__ Wt, int N,
    const float* __restrict__ bias,
    int mode, float* __restrict__ Ofp, __bf16* __restrict__ Obf,
    const float* __restrict__ res)
{
  __shared__ __align__(16) __bf16 As[2][128][40];
  __shared__ __align__(16) __bf16 Bs[2][64][40];
  const int tid = threadIdx.x, lane = tid & 31, w = tid >> 5;
  const int wm = w >> 1, wn = w & 1;
  const int half = lane >> 4, l15 = lane & 15;
  const int m0 = blockIdx.y * 128, n0 = blockIdx.x * 64;
  const int l4q = lane >> 2;          // 0..7  : row within 8-row strip
  const int koffl = (lane & 3) * 8;   // 0,8,16,24 : element offset in 32-wide K

  f32x8 c[2][2];
  c[0][0] = c[0][1] = c[1][0] = c[1][1] = (f32x8)0.0f;

  // One stage = 3 async wave-ops per wave: 16 A rows + 8 B rows.
  auto stage = [&](int buf, int k0) {
    #pragma unroll
    for (int i = 0; i < 2; ++i) {
      int rl = w * 16 + i * 8 + l4q;
      int ar = m0 + rl; if (ar > M - 1) ar = M - 1;   // clamp; guarded at store
      async_copy_b128(&As[buf][rl][koffl], A + (size_t)ar * Kd + k0 + koffl);
    }
    {
      int rl = w * 8 + l4q;
      async_copy_b128(&Bs[buf][rl][koffl], Wt + (size_t)(n0 + rl) * Kd + k0 + koffl);
    }
  };

  stage(0, 0);
  const int nk = Kd >> 5;
  for (int it = 0; it < nk; ++it) {
    const int cur = it & 1;
    if (it + 1 < nk) {
      stage(cur ^ 1, (it + 1) << 5);
      wait_async_le3();                // current tile's 3 ops retired (in-order)
    } else {
      wait_async_0();
    }
    __syncthreads();                   // all waves' DMA visible

    bf16x16 a[2], b[2];
    #pragma unroll
    for (int i = 0; i < 2; ++i)
      a[i] = ldfrag(&As[cur][wm*32 + i*16 + l15][half*8],
                    &As[cur][wm*32 + i*16 + l15][16 + half*8]);
    #pragma unroll
    for (int j = 0; j < 2; ++j)
      b[j] = ldfrag(&Bs[cur][wn*32 + j*16 + l15][half*8],
                    &Bs[cur][wn*32 + j*16 + l15][16 + half*8]);
    #pragma unroll
    for (int i = 0; i < 2; ++i)
      #pragma unroll
      for (int j = 0; j < 2; ++j)
        c[i][j] = wmma_bf16(a[i], b[j], c[i][j]);
    __syncthreads();                   // reads done before buffer is re-filled
  }

  #pragma unroll
  for (int i = 0; i < 2; ++i)
    #pragma unroll
    for (int j = 0; j < 2; ++j)
      #pragma unroll
      for (int r = 0; r < 8; ++r) {
        int row = m0 + wm*32 + i*16 + r + half*8;
        int col = n0 + wn*32 + j*16 + l15;
        if (row < M) {
          float v = c[i][j][r] + bias[col];
          size_t o = (size_t)row * N + col;
          if (mode == 0)      Ofp[o] = v;
          else if (mode == 1) Obf[o] = (__bf16)(0.5f * v * (1.0f + erff(v * 0.70710678118f)));
          else if (mode == 2) Ofp[o] = res[o] + v;
          else                Obf[o] = (__bf16)v;
        }
      }
}

// ---------------------------------------------------------------------------
// Segment attention: 1 block / segment, 1 wave / head. Rows padded 18->32.
// ---------------------------------------------------------------------------
__device__ __forceinline__ const __bf16* seg_row(const __bf16* Mtx, const __bf16* mark,
                                                 size_t segbase, int m) {
  if (m == 0)  return mark;
  if (m <= 16) return Mtx + (segbase + (size_t)(m - 1)) * Hdim;
  if (m == 17) return mark + Hdim;
  return nullptr;
}

__global__ __launch_bounds__(256) void k_attention(
    const __bf16* __restrict__ Qm, const __bf16* __restrict__ Km, const __bf16* __restrict__ Vm,
    const __bf16* __restrict__ mq, const __bf16* __restrict__ mk, const __bf16* __restrict__ mv,
    __bf16* __restrict__ AO)
{
  __shared__ __align__(16) __bf16 Pls[8][32][40];
  __shared__ __align__(16) __bf16 Vt[8][64][40];
  const int tid = threadIdx.x;
  const int lane = tid & 31;
  const int w = tid >> 5;                          // head
  const int half = lane >> 4;
  const int l15 = lane & 15;
  const int sidx = blockIdx.x;
  const size_t segbase = (size_t)(sidx >> 8) * CORE + (size_t)(sidx & 255) * SEGL;

  // stage V^T for this head: Vt[hd][key]
  {
    const __bf16* rp = seg_row(Vm, mv, segbase, lane);
    #pragma unroll 4
    for (int hd = 0; hd < HD; ++hd)
      Vt[w][hd][lane] = rp ? rp[w * HD + hd] : (__bf16)0.0f;
  }

  // S = Q K^T (A and B frags gathered straight from global, rows padded w/ zero)
  f32x8 s[2][2];
  s[0][0] = s[0][1] = s[1][0] = s[1][1] = (f32x8)0.0f;
  #pragma unroll
  for (int kc = 0; kc < 2; ++kc) {
    bf16x16 aq[2], bk[2];
    #pragma unroll
    for (int f = 0; f < 2; ++f) {
      int m = l15 + f * 16;
      const __bf16* rq = seg_row(Qm, mq, segbase, m);
      aq[f] = rq ? ldfrag(rq + w*HD + kc*32 + half*8, rq + w*HD + kc*32 + 16 + half*8) : zfrag();
      const __bf16* rk = seg_row(Km, mk, segbase, m);
      bk[f] = rk ? ldfrag(rk + w*HD + kc*32 + half*8, rk + w*HD + kc*32 + 16 + half*8) : zfrag();
    }
    #pragma unroll
    for (int i = 0; i < 2; ++i)
      #pragma unroll
      for (int j = 0; j < 2; ++j)
        s[i][j] = wmma_bf16(aq[i], bk[j], s[i][j]);
  }

  // masked softmax over keys 0..17 (scale 1/sqrt(64))
  #pragma unroll
  for (int fm = 0; fm < 2; ++fm) {
    #pragma unroll
    for (int r = 0; r < 8; ++r) {
      float v0 = s[fm][0][r] * 0.125f;             // cols 0..15
      float v1 = s[fm][1][r] * 0.125f;             // cols 16..31
      if (l15 >= 2) v1 = -1e30f;                   // cols >= 18 masked
      float mx = fmaxf(v0, v1);
      #pragma unroll
      for (int off = 8; off; off >>= 1) mx = fmaxf(mx, __shfl_xor(mx, off, 32));
      float e0 = __expf(v0 - mx);
      float e1 = __expf(v1 - mx);
      float sum = e0 + e1;
      #pragma unroll
      for (int off = 8; off; off >>= 1) sum += __shfl_xor(sum, off, 32);
      float inv = 1.0f / sum;
      int mrow = fm * 16 + r + half * 8;
      Pls[w][mrow][l15]      = (__bf16)(e0 * inv);
      Pls[w][mrow][16 + l15] = (__bf16)(e1 * inv);
    }
  }

  // O = P @ V
  bf16x16 ap[2], bv[4];
  #pragma unroll
  for (int fm = 0; fm < 2; ++fm)
    ap[fm] = ldfrag(&Pls[w][l15 + fm*16][half*8], &Pls[w][l15 + fm*16][16 + half*8]);
  #pragma unroll
  for (int fn = 0; fn < 4; ++fn)
    bv[fn] = ldfrag(&Vt[w][fn*16 + l15][half*8], &Vt[w][fn*16 + l15][16 + half*8]);
  f32x8 o[2][4];
  #pragma unroll
  for (int fm = 0; fm < 2; ++fm)
    #pragma unroll
    for (int fn = 0; fn < 4; ++fn) {
      o[fm][fn] = (f32x8)0.0f;
      o[fm][fn] = wmma_bf16(ap[fm], bv[fn], o[fm][fn]);
    }

  #pragma unroll
  for (int fm = 0; fm < 2; ++fm)
    #pragma unroll
    for (int fn = 0; fn < 4; ++fn)
      #pragma unroll
      for (int r = 0; r < 8; ++r) {
        int mrow = fm * 16 + r + half * 8;
        if (mrow >= 1 && mrow <= 16) {
          size_t tok = segbase + (size_t)(mrow - 1);
          AO[tok * Hdim + (size_t)(w * HD + fn * 16 + l15)] = (__bf16)o[fm][fn][r];
        }
      }
}

// ---------------------------------------------------------------------------
// LayerNorm over H=512, one wave per row; write bf16 FFN input (row remapped)
// ---------------------------------------------------------------------------
__global__ __launch_bounds__(256) void k_layernorm(
    const float* __restrict__ O, const float* __restrict__ g, const float* __restrict__ bb,
    __bf16* __restrict__ X)
{
  const int lane = threadIdx.x & 31, w = threadIdx.x >> 5;
  const int row = blockIdx.x * 8 + w;             // 65536 rows
  const float* p = O + (size_t)row * Hdim;
  float vals[16], s = 0.0f, s2 = 0.0f;
  #pragma unroll
  for (int i = 0; i < 16; ++i) {
    float v = p[lane + i * 32];
    vals[i] = v; s += v; s2 += v * v;
  }
  #pragma unroll
  for (int off = 16; off; off >>= 1) { s += __shfl_xor(s, off, 32); s2 += __shfl_xor(s2, off, 32); }
  float mean = s * (1.0f / Hdim);
  float var  = s2 * (1.0f / Hdim) - mean * mean;
  float inv  = rsqrtf(var + 1e-5f);
  const size_t xrow = (size_t)row + (size_t)(row >> 12);  // b*4097 + i
  __bf16* q = X + xrow * Hdim;
  #pragma unroll
  for (int i = 0; i < 16; ++i) {
    int col = lane + i * 32;
    q[col] = (__bf16)((vals[i] - mean) * inv * g[col] + bb[col]);
  }
}

// last token of each batch: FFN input = original hidden
__global__ void k_fill_last(const float* __restrict__ hidden, __bf16* __restrict__ X) {
  int idx = blockIdx.x * 256 + threadIdx.x;       // 16*512
  if (idx >= 16 * Hdim) return;
  int b = idx >> 9, col = idx & 511;
  size_t hrow = (size_t)b * 4097 + 4096;
  X[hrow * Hdim + col] = (__bf16)hidden[hrow * Hdim + col];
}

// ---------------------------------------------------------------------------
extern "C" void kernel_launch(void* const* d_in, const int* in_sizes, int n_in,
                              void* d_out, int out_size, void* d_ws, size_t ws_size,
                              hipStream_t stream)
{
  (void)in_sizes; (void)n_in; (void)out_size; (void)ws_size;
  const float* hidden = (const float*)d_in[0];
  const float* sstart = (const float*)d_in[3];
  const float* send   = (const float*)d_in[4];
  const float* Wq = (const float*)d_in[5];  const float* bq = (const float*)d_in[6];
  const float* Wk = (const float*)d_in[7];  const float* bk = (const float*)d_in[8];
  const float* Wv = (const float*)d_in[9];  const float* bv = (const float*)d_in[10];
  const float* Wo = (const float*)d_in[11]; const float* bo = (const float*)d_in[12];
  const float* lng = (const float*)d_in[13]; const float* lnb = (const float*)d_in[14];
  const float* W1 = (const float*)d_in[15]; const float* b1 = (const float*)d_in[16];
  const float* W2 = (const float*)d_in[17]; const float* b2 = (const float*)d_in[18];
  float* out = (float*)d_out;

  char* ws = (char*)d_ws;
  // Region plan (phase reuse):
  //   XC (bf16 core hidden) : dead after QKV     -> reused by Xffn
  //   Q/K/V                 : dead after attention-> O (fp32) reuses Q.., Hmid reuses K..
  //   ATT                   : dead after out-proj
  const size_t OXC = 0;
  const size_t OQ  = (size_t)0x04400000;   //  68 MB
  const size_t OK_ = (size_t)0x08800000;   // 136 MB
  const size_t OV  = (size_t)0x0CC00000;   // 204 MB
  const size_t OAT = (size_t)0x11000000;   // 272 MB
  const size_t OW  = (size_t)0x15400000;   // 340 MB (weights + markers)
  __bf16* XC   = (__bf16*)(ws + OXC);
  __bf16* Qb   = (__bf16*)(ws + OQ);
  __bf16* Kb   = (__bf16*)(ws + OK_);
  __bf16* Vb   = (__bf16*)(ws + OV);
  __bf16* AT   = (__bf16*)(ws + OAT);
  float*  Obuf = (float*)(ws + OQ);        // fp32 attn-proj output (reuses Q+K)
  __bf16* Xf   = (__bf16*)(ws + OXC);      // FFN input (reuses XC)
  __bf16* Hm   = (__bf16*)(ws + OK_);      // FFN hidden (reuses K..V span)
  __bf16* Wqt  = (__bf16*)(ws + OW);
  __bf16* Wkt  = (__bf16*)(ws + OW + 0x080000);
  __bf16* Wvt  = (__bf16*)(ws + OW + 0x100000);
  __bf16* Wot  = (__bf16*)(ws + OW + 0x180000);
  __bf16* W1t  = (__bf16*)(ws + OW + 0x200000);
  __bf16* W2t  = (__bf16*)(ws + OW + 0x300000);
  __bf16* mq   = (__bf16*)(ws + OW + 0x400000);
  __bf16* mk   = (__bf16*)(ws + OW + 0x400800);
  __bf16* mv   = (__bf16*)(ws + OW + 0x401000);

  // 1. weights -> bf16 transposed, hidden core -> bf16
  k_transpose_w<<<1024, 256, 0, stream>>>(Wq, Wqt, 512, 512);
  k_transpose_w<<<1024, 256, 0, stream>>>(Wk, Wkt, 512, 512);
  k_transpose_w<<<1024, 256, 0, stream>>>(Wv, Wvt, 512, 512);
  k_transpose_w<<<1024, 256, 0, stream>>>(Wo, Wot, 512, 512);
  k_transpose_w<<<2048, 256, 0, stream>>>(W1, W1t, 512, 1024);
  k_transpose_w<<<2048, 256, 0, stream>>>(W2, W2t, 1024, 512);
  k_convert_hidden<<<32768, 256, 0, stream>>>(hidden, XC);
  // 2. marker-token QKV (shared by every segment)
  k_marker_qkv<<<12, 256, 0, stream>>>(sstart, send, Wq, bq, Wk, bk, Wv, bv, mq, mk, mv);
  // 3. QKV projections over 65536 core tokens
  dim3 gq(8, 512);
  k_gemm_bf16<<<gq, 256, 0, stream>>>(XC, MTOK, 512, Wqt, 512, bq, 3, nullptr, Qb, nullptr);
  k_gemm_bf16<<<gq, 256, 0, stream>>>(XC, MTOK, 512, Wkt, 512, bk, 3, nullptr, Kb, nullptr);
  k_gemm_bf16<<<gq, 256, 0, stream>>>(XC, MTOK, 512, Wvt, 512, bv, 3, nullptr, Vb, nullptr);
  // 4. segment attention (4096 segments x 8 heads)
  k_attention<<<4096, 256, 0, stream>>>(Qb, Kb, Vb, mq, mk, mv, AT);
  // 5. output projection -> fp32
  k_gemm_bf16<<<dim3(8, 512), 256, 0, stream>>>(AT, MTOK, 512, Wot, 512, bo,
                                                0, Obuf, nullptr, nullptr);
  // 6. LayerNorm -> bf16 FFN input (b*4097+s layout)
  k_layernorm<<<8192, 256, 0, stream>>>(Obuf, lng, lnb, Xf);
  // 7. trailing token per batch passes through
  k_fill_last<<<32, 256, 0, stream>>>(hidden, Xf);
  // 8. FFN1 with exact GELU -> bf16
  k_gemm_bf16<<<dim3(16, 513), 256, 0, stream>>>(Xf, MTOT, 512, W1t, 1024, b1,
                                                 1, nullptr, Hm, nullptr);
  // 9. FFN2 + residual(original hidden) -> fp32 d_out
  k_gemm_bf16<<<dim3(8, 513), 256, 0, stream>>>(Hm, MTOT, 1024, W2t, 512, b2,
                                                2, out, nullptr, hidden);
}
